// GCN_20847771254904
// MI455X (gfx1250) — matrix operand order repaired and checked by reading, not compiled, not run on hardware
//
#include <hip/hip_runtime.h>
#include <stdint.h>

// ---------------------------------------------------------------------------
// GCN on gfx1250: bf16x3-split WMMA GEMMs (pre-packed fragment layout) +
// L2-resident atomic edge aggregation.
// ---------------------------------------------------------------------------

typedef __attribute__((ext_vector_type(16))) __bf16 v16bf;
typedef __attribute__((ext_vector_type(8)))  float  v8f;

#define KDIM 128            // feature width into every GEMM (F_IN == H == 128)
#define WAVE 32
#define WPB  8              // waves per block
#define TPB  (WPB * WAVE)

union BfPack { unsigned u[8]; v16bf v; };

// Split fp32 into bf16 hi (RNE) + bf16 lo (residual, RNE).
static __device__ __forceinline__ void split_bf16(float a, unsigned& h, unsigned& l) {
    unsigned ua = __float_as_uint(a);
    unsigned hr = (ua + (0x7FFFu + ((ua >> 16) & 1u))) & 0xFFFF0000u;
    h = hr >> 16;
    float lf = a - __uint_as_float(hr);
    unsigned ul = __float_as_uint(lf);
    l = (ul + (0x7FFFu + ((ul >> 16) & 1u))) >> 16;
}

// ------------------------------- degree / norm ------------------------------

__global__ void k_init_deg(float* __restrict__ deg, int n) {
    int i = blockIdx.x * blockDim.x + threadIdx.x;
    if (i < n) deg[i] = 1.0f;                         // self-loop
}

__global__ void k_prep_edges(const long long* __restrict__ eidx,
                             int* __restrict__ src, int* __restrict__ dst,
                             float* __restrict__ deg, int E) {
    int e = blockIdx.x * blockDim.x + threadIdx.x;
    if (e >= E) return;
    int s = (int)eidx[e];
    int d = (int)eidx[(size_t)E + e];
    src[e] = s;
    dst[e] = d;
    atomicAdd(&deg[d], 1.0f);
}

__global__ void k_deg_to_dinv(float* __restrict__ deg, int n) {
    int i = blockIdx.x * blockDim.x + threadIdx.x;
    if (i < n) deg[i] = rsqrtf(deg[i]);               // deg >= 1 always
}

// ------------------------- weight split + pack (once) -----------------------
// W is K x M row-major fp32.  Output: K x M/2 uint32, [15:0]=col even bf16,
// [31:16]=col odd bf16  -> GEMM re-pairs along k into B VGPRs.

__global__ void k_convert_w(const float* __restrict__ W,
                            unsigned* __restrict__ hi, unsigned* __restrict__ lo,
                            int pairs) {
    int i = blockIdx.x * blockDim.x + threadIdx.x;
    if (i >= pairs) return;
    float a = W[2 * i], b = W[2 * i + 1];
    unsigned ha, la, hb, lb;
    split_bf16(a, ha, la);
    split_bf16(b, hb, lb);
    hi[i] = ha | (hb << 16);
    lo[i] = la | (lb << 16);
}

// --------------------- activation pack into A fragments ---------------------
// Reads fp32 activations (nrows x 128), optionally applies swish, splits to
// bf16 hi/lo, and stores in WMMA A-fragment order so the GEMM lane fetch is
// two b128 loads:  dword index = ((rt*4 + ks)*32 + lane)*8 + v
//   where for k-offset ko in [0,32): pair=ko&1, v = 4*(ko>>4) + ((ko>>1)&3),
//   lane = (row&15) + 16*((ko>>3)&1).

__global__ void k_pack_a(const float* __restrict__ act,
                         unsigned* __restrict__ fh, unsigned* __restrict__ fl,
                         int nrows, int applySwish) {
    int t = blockIdx.x * blockDim.x + threadIdx.x;    // one per bf16-pair dword
    if (t >= nrows * 64) return;
    const int r  = t >> 6;
    const int k  = (t & 63) << 1;
    float a = act[(size_t)r * KDIM + k];
    float b = act[(size_t)r * KDIM + k + 1];
    if (applySwish) {
        a = a / (1.0f + expf(-a));
        b = b / (1.0f + expf(-b));
    }
    unsigned ha, la, hb, lb;
    split_bf16(a, ha, la);
    split_bf16(b, hb, lb);
    const int rt   = r >> 4;
    const int ks   = k >> 5;
    const int ko   = k & 31;
    const int lane = (r & 15) + (((ko >> 3) & 1) << 4);
    const int v    = ((ko >> 4) << 2) + ((ko >> 1) & 3);
    const size_t idx = (((size_t)(rt * 4 + ks) * 32) + lane) * 8 + v;
    fh[idx] = ha | (hb << 16);
    fl[idx] = la | (lb << 16);
}

// ------------------------------- WMMA GEMM ----------------------------------
// Out[N x M] = A[N x 128] @ W[128 x M].  A pre-packed in fragment order
// (hi/lo), W packed as bf16 column pairs (hi/lo).  One wave computes a 16x64
// output strip (4 accumulators); bf16x3 per k-step: Ah*Bh + Ah*Bl + Al*Bh.

__global__ void __launch_bounds__(TPB)
k_gemm_packed(const unsigned* __restrict__ fh, const unsigned* __restrict__ fl,
              const unsigned* __restrict__ Whi, const unsigned* __restrict__ Wlo,
              float* __restrict__ Out, int nrows, int M) {
    const int mblocks = M >> 6;                       // 64-wide strips
    const int wave  = (blockIdx.x * blockDim.x + threadIdx.x) >> 5;
    const int total = (nrows >> 4) * mblocks;
    if (wave >= total) return;                        // wave-uniform: EXEC full

    const int lane = threadIdx.x & 31;
    const int rt   = wave / mblocks;
    const int mb   = wave - rt * mblocks;
    const int n0   = mb << 6;
    const int mw   = M >> 1;                          // packed row width (dwords)

    v8f acc[4] = {{}, {}, {}, {}};

    #pragma unroll
    for (int ks = 0; ks < 4; ++ks) {
        // A fragments: lane-contiguous, two b128 loads each for hi and lo
        const size_t fidx = (((size_t)(rt * 4 + ks) * 32) + lane) * 8;
        BfPack ah, al;
        *(uint4*)(&ah.u[0]) = *(const uint4*)(fh + fidx);
        *(uint4*)(&ah.u[4]) = *(const uint4*)(fh + fidx + 4);
        *(uint4*)(&al.u[0]) = *(const uint4*)(fl + fidx);
        *(uint4*)(&al.u[4]) = *(const uint4*)(fl + fidx + 4);

        // B: lane l holds column n = l&15, k = 8*(l>>4) + s (s<8), 16 + 8*(l>>4) + (s-8) (s>=8)
        const unsigned* bh = Whi + (size_t)(ks * 32 + ((lane >> 4) << 3)) * mw + (n0 >> 1);
        const unsigned* bl = Wlo + (size_t)(ks * 32 + ((lane >> 4) << 3)) * mw + (n0 >> 1);
        #pragma unroll
        for (int tl = 0; tl < 4; ++tl) {
            BfPack Bh, Bl;
            const int cc = tl * 16 + (lane & 15);
            const int hs = (cc & 1) << 4;
            #pragma unroll
            for (int v = 0; v < 8; ++v) {
                const int kr = ((v >> 2) << 4) + ((v & 3) << 1);
                const unsigned* ph = bh + (size_t)kr * mw + (cc >> 1);
                const unsigned* pl = bl + (size_t)kr * mw + (cc >> 1);
                Bh.u[v] = ((ph[0] >> hs) & 0xFFFFu) | (((ph[mw] >> hs) & 0xFFFFu) << 16);
                Bl.u[v] = ((pl[0] >> hs) & 0xFFFFu) | (((pl[mw] >> hs) & 0xFFFFu) << 16);
            }
            acc[tl] = __builtin_amdgcn_wmma_f32_16x16x32_bf16(false, ah.v, false, Bh.v,
                                                              (short)0, acc[tl], false, false);
            acc[tl] = __builtin_amdgcn_wmma_f32_16x16x32_bf16(false, ah.v, false, Bl.v,
                                                              (short)0, acc[tl], false, false);
            acc[tl] = __builtin_amdgcn_wmma_f32_16x16x32_bf16(false, al.v, false, Bh.v,
                                                              (short)0, acc[tl], false, false);
        }
    }

    // D store: VGPR v holds row (v + 8*(lane>=16)), col = lane&15 within tile
    const int row0 = (rt << 4) + ((lane >> 4) << 3);
    const int col  = lane & 15;
    #pragma unroll
    for (int tl = 0; tl < 4; ++tl) {
        #pragma unroll
        for (int v = 0; v < 8; ++v) {
            Out[(size_t)(row0 + v) * M + n0 + tl * 16 + col] = acc[tl][v];
        }
    }
}

// ------------------------------ aggregation ---------------------------------
// out[i,:] = dinv[i]^2 * h[i,:] + b[:]   (self-loop term + bias; inits out)

__global__ void k_agg_init(const float* __restrict__ h, const float* __restrict__ dinv,
                           const float* __restrict__ b, float* __restrict__ out,
                           int n, int M) {
    const int per = M >> 2;
    const int idx = blockIdx.x * blockDim.x + threadIdx.x;
    if (idx >= n * per) return;
    const int i = idx / per;
    const int f = (idx - i * per) << 2;
    const float di = dinv[i];
    const float sc = di * di;
    const float4 hv = *(const float4*)(h + (size_t)i * M + f);
    const float4 bv = *(const float4*)(b + f);
    float4 o;
    o.x = sc * hv.x + bv.x;
    o.y = sc * hv.y + bv.y;
    o.z = sc * hv.z + bv.z;
    o.w = sc * hv.w + bv.w;
    *(float4*)(out + (size_t)i * M + f) = o;
}

// One wave per edge: gather h[src] (L2-resident), scatter-add norm*h into out[dst].
template <int MM>
__global__ void __launch_bounds__(TPB)
k_agg_edge(const float* __restrict__ h, const int* __restrict__ src,
           const int* __restrict__ dst, const float* __restrict__ dinv,
           float* __restrict__ out, int E) {
    const int wave = (blockIdx.x * blockDim.x + threadIdx.x) >> 5;
    if (wave >= E) return;
    const int lane = threadIdx.x & 31;
    const int s = src[wave];
    const int d = dst[wave];
    const float nrm = dinv[s] * dinv[d];
    if constexpr (MM == 128) {
        const float4 v = *(const float4*)(h + (size_t)s * 128 + lane * 4);
        float* o = out + (size_t)d * 128 + lane * 4;
        atomicAdd(o + 0, nrm * v.x);
        atomicAdd(o + 1, nrm * v.y);
        atomicAdd(o + 2, nrm * v.z);
        atomicAdd(o + 3, nrm * v.w);
    } else {  // MM == 64
        const float2 v = *(const float2*)(h + (size_t)s * 64 + lane * 2);
        float* o = out + (size_t)d * 64 + lane * 2;
        atomicAdd(o + 0, nrm * v.x);
        atomicAdd(o + 1, nrm * v.y);
    }
}

// ------------------------------ log-softmax ---------------------------------
// Over C=64 classes; one wave per row, 2 elems per lane (wave32).

__global__ void __launch_bounds__(TPB)
k_log_softmax64(float* __restrict__ io, int nrows) {
    const int wave = (blockIdx.x * blockDim.x + threadIdx.x) >> 5;
    if (wave >= nrows) return;
    const int lane = threadIdx.x & 31;
    float2 v = *(float2*)(io + (size_t)wave * 64 + lane * 2);
    float m = fmaxf(v.x, v.y);
    #pragma unroll
    for (int o = 16; o > 0; o >>= 1) m = fmaxf(m, __shfl_xor(m, o, 32));
    float s = expf(v.x - m) + expf(v.y - m);
    #pragma unroll
    for (int o = 16; o > 0; o >>= 1) s += __shfl_xor(s, o, 32);
    const float lse = m + logf(s);
    v.x -= lse;
    v.y -= lse;
    *(float2*)(io + (size_t)wave * 64 + lane * 2) = v;
}

// ------------------------------- launcher -----------------------------------

static inline int cdiv_i(long long a, int b) { return (int)((a + b - 1) / b); }

extern "C" void kernel_launch(void* const* d_in, const int* in_sizes, int n_in,
                              void* d_out, int out_size, void* d_ws, size_t ws_size,
                              hipStream_t stream) {
    const float*     x    = (const float*)d_in[0];
    const long long* eidx = (const long long*)d_in[1];
    const float*     W0   = (const float*)d_in[2];
    const float*     b0   = (const float*)d_in[3];
    const float*     W1   = (const float*)d_in[4];
    const float*     b1   = (const float*)d_in[5];
    const float*     W2   = (const float*)d_in[6];
    const float*     b2   = (const float*)d_in[7];

    const int N = in_sizes[0] / KDIM;   // 50000
    const int E = in_sizes[1] / 2;      // 800000
    const int H = 128;
    const int C = in_sizes[7];          // 64

    // workspace carve-out (256B aligned)
    char*  ws  = (char*)d_ws;
    size_t off = 0;
    auto carve = [&](size_t bytes) -> void* {
        void* p = ws + off;
        off = (off + bytes + 255) & ~(size_t)255;
        return p;
    };
    float*    deg  = (float*)carve((size_t)N * 4);        // becomes dinv in place
    int*      src  = (int*)carve((size_t)E * 4);
    int*      dst  = (int*)carve((size_t)E * 4);
    unsigned* w0h  = (unsigned*)carve((size_t)KDIM * (H / 2) * 4);
    unsigned* w0l  = (unsigned*)carve((size_t)KDIM * (H / 2) * 4);
    unsigned* w1h  = (unsigned*)carve((size_t)KDIM * (H / 2) * 4);
    unsigned* w1l  = (unsigned*)carve((size_t)KDIM * (H / 2) * 4);
    unsigned* w2h  = (unsigned*)carve((size_t)KDIM * (C / 2) * 4);
    unsigned* w2l  = (unsigned*)carve((size_t)KDIM * (C / 2) * 4);
    unsigned* fh   = (unsigned*)carve((size_t)N * 64 * 4);   // A-frag hi
    unsigned* fl   = (unsigned*)carve((size_t)N * 64 * 4);   // A-frag lo
    float*    bufA = (float*)carve((size_t)N * H * 4);
    float*    bufB = (float*)carve((size_t)N * H * 4);
    (void)ws_size;

    float* outF = (float*)d_out;

    // --- normalization ---
    k_init_deg<<<cdiv_i(N, 256), 256, 0, stream>>>(deg, N);
    k_prep_edges<<<cdiv_i(E, 256), 256, 0, stream>>>(eidx, src, dst, deg, E);
    k_deg_to_dinv<<<cdiv_i(N, 256), 256, 0, stream>>>(deg, N);
    float* dinv = deg;

    // --- weight split/pack ---
    k_convert_w<<<cdiv_i((long long)KDIM * H / 2, 256), 256, 0, stream>>>(W0, w0h, w0l, KDIM * H / 2);
    k_convert_w<<<cdiv_i((long long)KDIM * H / 2, 256), 256, 0, stream>>>(W1, w1h, w1l, KDIM * H / 2);
    k_convert_w<<<cdiv_i((long long)KDIM * C / 2, 256), 256, 0, stream>>>(W2, w2h, w2l, KDIM * C / 2);

    const int packBlocks = cdiv_i((long long)N * 64, 256);
    const int gemmBlksH  = cdiv_i((N / 16) * (H / 64), WPB);
    const int gemmBlksC  = cdiv_i((N / 16) * (C / 64), WPB);
    const int edgeBlocks = cdiv_i(E, WPB);

    // --- layer 1: x -> bufA -> bufB ---
    k_pack_a<<<packBlocks, 256, 0, stream>>>(x, fh, fl, N, 0);
    k_gemm_packed<<<gemmBlksH, TPB, 0, stream>>>(fh, fl, w0h, w0l, bufA, N, H);
    k_agg_init<<<cdiv_i((long long)N * H / 4, 256), 256, 0, stream>>>(bufA, dinv, b0, bufB, N, H);
    k_agg_edge<128><<<edgeBlocks, TPB, 0, stream>>>(bufA, src, dst, dinv, bufB, E);

    // --- layer 2: bufB -(swish in pack)-> bufA -> bufB ---
    k_pack_a<<<packBlocks, 256, 0, stream>>>(bufB, fh, fl, N, 1);
    k_gemm_packed<<<gemmBlksH, TPB, 0, stream>>>(fh, fl, w1h, w1l, bufA, N, H);
    k_agg_init<<<cdiv_i((long long)N * H / 4, 256), 256, 0, stream>>>(bufA, dinv, b1, bufB, N, H);
    k_agg_edge<128><<<edgeBlocks, TPB, 0, stream>>>(bufA, src, dst, dinv, bufB, E);

    // --- layer 3: bufB -(swish in pack)-> bufA(N x 64) -> d_out ---
    k_pack_a<<<packBlocks, 256, 0, stream>>>(bufB, fh, fl, N, 1);
    k_gemm_packed<<<gemmBlksC, TPB, 0, stream>>>(fh, fl, w2h, w2l, bufA, N, C);
    k_agg_init<<<cdiv_i((long long)N * C / 4, 256), 256, 0, stream>>>(bufA, dinv, b2, outF, N, C);
    k_agg_edge<64><<<edgeBlocks, TPB, 0, stream>>>(bufA, src, dst, dinv, outF, E);
    k_log_softmax64<<<cdiv_i(N, WPB), TPB, 0, stream>>>(outF, N);
}
